// DCINeuralODE_28209345200389
// MI455X (gfx1250) — compile-verified, matrix operand order checked by reading, not run on hardware
//
#include <hip/hip_runtime.h>
#include <hip/hip_bf16.h>

typedef __attribute__((ext_vector_type(2)))  float    v2f;
typedef __attribute__((ext_vector_type(8)))  float    v8f;
typedef __attribute__((ext_vector_type(16))) _Float16 v16h;

#define HIDP 128
#define HIDR 64
#define SEQ_H 1024
#define EPSV 1e-6f

__device__ __forceinline__ float rcp_fast(float x) {
#if __has_builtin(__builtin_amdgcn_rcpf)
  return __builtin_amdgcn_rcpf(x);
#else
  return 1.0f / x;
#endif
}

__device__ __forceinline__ float fast_tanh(float x) {
#if __has_builtin(__builtin_amdgcn_tanhf)
  return __builtin_amdgcn_tanhf(x);            // native v_tanh_f32 on gfx1250
#else
  float e = __expf(-2.0f * fabsf(x));          // tanh(|x|) = (1-e)/(1+e)
  float t = (1.0f - e) * rcp_fast(1.0f + e);
  return copysignf(t, x);
#endif
}

__device__ __forceinline__ float softplus_f(float x) {
  // log(1+e^x) = max(x,0) + log(1+e^-|x|)   (stable)
  return fmaxf(x, 0.0f) + __logf(1.0f + __expf(-fabsf(x)));
}

__global__ __launch_bounds__(32)
void dci_rollout_kernel(const float* __restrict__ I,   const float* __restrict__ Tz,
                        const float* __restrict__ soc0,
                        const float* __restrict__ W1p, const float* __restrict__ b1p,
                        const float* __restrict__ W2p, const float* __restrict__ b2p,
                        const float* __restrict__ W1r, const float* __restrict__ b1r,
                        const float* __restrict__ W2r, const float* __restrict__ b2r,
                        float* __restrict__ out)
{
  const int  lane = threadIdx.x;        // wave32: one wave per block
  const bool lo   = lane < 16;
  const int  m    = lane & 15;
  const int  row0 = blockIdx.x * 16;

  // ---------------- preload loop-invariant weight operands ----------------
  // Layer-1 A-tiles: W1^T augmented with bias as K=3 column (f32 16x4 A layout:
  //   lanes 0-15 hold K=0,1 ; lanes 16-31 hold K=2,3).
  v2f A1p[8];
#pragma unroll
  for (int j = 0; j < 8; ++j) {
    int col = 16 * j + m;
    A1p[j].x = lo ? W1p[0 * HIDP + col] : W1p[2 * HIDP + col]; // K=0: W1[0], K=2: W1[2]
    A1p[j].y = lo ? W1p[1 * HIDP + col] : b1p[col];            // K=1: W1[1], K=3: bias
  }
  v2f A1r[4];
#pragma unroll
  for (int j = 0; j < 4; ++j) {
    int col = 16 * j + m;
    A1r[j].x = lo ? W1r[0 * HIDR + col] : W1r[2 * HIDR + col];
    A1r[j].y = lo ? W1r[1 * HIDR + col] : b1r[col];
  }

  // Layer-2 A-tiles: W2^T as f16 16x32 A layout.
  // v16h component h maps to K = h + (h>=8 ? 8 : 0) + (hi-lane ? 8 : 0).
  v16h A2p[4];
#pragma unroll
  for (int c = 0; c < 4; ++c)
#pragma unroll
    for (int h = 0; h < 16; ++h) {
      int k = 32 * c + h + (h >= 8 ? 8 : 0) + (lo ? 0 : 8);
      A2p[c][h] = (_Float16)((m < 4) ? W2p[k * 4 + m] : 0.0f);
    }
  v16h A2r[2];
#pragma unroll
  for (int c = 0; c < 2; ++c)
#pragma unroll
    for (int h = 0; h < 16; ++h) {
      int k = 32 * c + h + (h >= 8 ? 8 : 0) + (lo ? 0 : 8);
      A2r[c][h] = (_Float16)((m == 0) ? W2r[k] : 0.0f);
    }

  // C-init for layer-2: bias broadcast (component r = param index).
  v8f Cp0 = {};
  Cp0[0] = b2p[0]; Cp0[1] = b2p[1]; Cp0[2] = b2p[2]; Cp0[3] = b2p[3];
  v8f Cr0 = {};
  Cr0[0] = b2r[0];

  // Per-lane streaming input: lanes 0-15 read I[row], lanes 16-31 read T[row].
  const float* srcIT = lo ? (I  + (size_t)(row0 + m) * SEQ_H)
                          : (Tz + (size_t)(row0 + m) * SEQ_H);
  float* outp = out + (size_t)(row0 + m) * SEQ_H;

  float s0  = soc0[row0 + m];
  float soc = (s0 != s0) ? 0.8f : s0;   // NaN -> 0.8
  float v1  = 0.0f;

  const v8f zeroC = {};

  // Software pipeline: `it` holds srcIT[t]; the load for t+1 is issued at the
  // top of the body so its latency is hidden behind the step's WMMA/tanh work.
  float it = srcIT[0];

#pragma unroll 1
  for (int t = 0; t < SEQ_H; ++t) {
    int tn = (t + 1 < SEQ_H) ? (t + 1) : t;
    float it_next = srcIT[tn];                  // issued now, consumed at bottom
    if ((t & 31) == 0)
      __builtin_prefetch(srcIT + t + 32, 0, 3); // next 128B line of the stream

    // feats^T B operand (f32 4x16 B layout, bias-row K=3 = 1.0)
    v2f Bf;
    Bf.x = lo ? soc : it;     // K=0: soc   | K=2: T
    Bf.y = lo ? it  : 1.0f;   // K=1: I     | K=3: 1

    // ---------- layer 1, both networks: 12 independent WMMAs ----------
    v8f Dp[8], Dr[4];
#pragma unroll
    for (int j = 0; j < 8; ++j)
      Dp[j] = __builtin_amdgcn_wmma_f32_16x16x4_f32(
                  false, A1p[j], false, Bf, (short)0, zeroC, false, false);
#pragma unroll
    for (int j = 0; j < 4; ++j)
      Dr[j] = __builtin_amdgcn_wmma_f32_16x16x4_f32(
                  false, A1r[j], false, Bf, (short)0, zeroC, false, false);

    // ---------- 96 independent tanh ----------
#pragma unroll
    for (int j = 0; j < 8; ++j)
#pragma unroll
      for (int r = 0; r < 8; ++r) Dp[j][r] = fast_tanh(Dp[j][r]);
#pragma unroll
    for (int j = 0; j < 4; ++j)
#pragma unroll
      for (int r = 0; r < 8; ++r) Dr[j][r] = fast_tanh(Dr[j][r]);

    // ---------- layer 2: lane-local f32->f16 repack + chained WMMAs ----------
    v8f accP = Cp0;
#pragma unroll
    for (int c = 0; c < 4; ++c) {
      v16h Bh;
#pragma unroll
      for (int h = 0; h < 16; ++h)
        Bh[h] = (_Float16)(h < 8 ? Dp[2 * c][h] : Dp[2 * c + 1][h - 8]);
      accP = __builtin_amdgcn_wmma_f32_16x16x32_f16(
                 false, A2p[c], false, Bh, (short)0, accP, false, false);
    }
    v8f accR = Cr0;
#pragma unroll
    for (int c = 0; c < 2; ++c) {
      v16h Bh;
#pragma unroll
      for (int h = 0; h < 16; ++h)
        Bh[h] = (_Float16)(h < 8 ? Dr[2 * c][h] : Dr[2 * c + 1][h - 8]);
      accR = __builtin_amdgcn_wmma_f32_16x16x32_f16(
                 false, A2r[c], false, Bh, (short)0, accR, false, false);
    }

    // ---------- per-lane ODE update (batch row = lane, lanes 0-15) ----------
    float R0 = softplus_f(accP[0]) * 0.01f   + EPSV;
    float R1 = softplus_f(accP[1]) * 0.02f   + EPSV;
    float C1 = softplus_f(accP[2]) * 2000.0f + EPSV;
    float Q  = softplus_f(accP[3]) * 5.0f    + EPSV;
    float resid = accR[0];

    float invC1 = rcp_fast(C1);
    float v1n   = v1 + (-v1 * rcp_fast(R1 * C1) + it * invC1);          // DT = 1
    float socn  = fminf(fmaxf(soc - it * rcp_fast(Q * 3600.0f), 0.0f), 1.0f);
    float ocv   = 3.0f + socn * (1.2f + socn * (-0.5f + 0.3f * socn));
    float vpred = ocv - it * R0 - v1n + resid;

    soc = socn;
    v1  = v1n;
    if (lo) outp[t] = vpred;

    it = it_next;   // consume pipelined load here (wait sinks to this point)
  }
}

extern "C" void kernel_launch(void* const* d_in, const int* in_sizes, int n_in,
                              void* d_out, int out_size, void* d_ws, size_t ws_size,
                              hipStream_t stream) {
  // setup_inputs order: V, I, Tz, soc0, W1p, b1p, W2p, b2p, W1r, b1r, W2r, b2r
  const float* I    = (const float*)d_in[1];
  const float* Tz   = (const float*)d_in[2];
  const float* soc0 = (const float*)d_in[3];
  const float* W1p  = (const float*)d_in[4];
  const float* b1p  = (const float*)d_in[5];
  const float* W2p  = (const float*)d_in[6];
  const float* b2p  = (const float*)d_in[7];
  const float* W1r  = (const float*)d_in[8];
  const float* b1r  = (const float*)d_in[9];
  const float* W2r  = (const float*)d_in[10];
  const float* b2r  = (const float*)d_in[11];
  float* out = (float*)d_out;

  dim3 grid(64);   // 1024 batch rows / 16 per wave
  dim3 block(32);  // one wave32 per workgroup
  hipLaunchKernelGGL(dci_rollout_kernel, grid, block, 0, stream,
                     I, Tz, soc0, W1p, b1p, W2p, b2p, W1r, b1r, W2r, b2r, out);
}
